// DynamicReductionNetwork_25683904430200
// MI455X (gfx1250) — matrix operand order repaired and checked by reading, not compiled, not run on hardware
//
#include <hip/hip_runtime.h>
#include <hip/hip_bf16.h>

// ---------------------------------------------------------------------------
// DynamicReductionNetwork on MI455X (gfx1250, wave32, WMMA)
// N=8192 nodes, B=4 batches, K=16 neighbors, H=64 hidden, IN=5
// ---------------------------------------------------------------------------

#define NN     8192
#define NB     4
#define KK     16
#define HH     64
#define NIN    5
#define NE     (2 * NN * KK)   // 262144 edges
#define INF_F  __builtin_inff()

typedef _Float16 v16h __attribute__((ext_vector_type(16)));
typedef float    v8f  __attribute__((ext_vector_type(8)));
typedef int      v4i_t __attribute__((vector_size(16)));   // matches builtin proto

#if __has_builtin(__builtin_amdgcn_global_load_async_to_lds_b128) && \
    __has_builtin(__builtin_amdgcn_s_wait_asynccnt)
#define HAVE_ASYNC_LDS 1
#else
#define HAVE_ASYNC_LDS 0
#endif

__device__ __forceinline__ float eluf(float x) {
    return x > 0.0f ? x : __expf(x) - 1.0f;
}
// monotone float->u32 encoding so atomicMax(u32) == float max (handles -inf)
__device__ __forceinline__ unsigned encf(float f) {
    unsigned u = __float_as_uint(f);
    return (u & 0x80000000u) ? ~u : (u | 0x80000000u);
}
__device__ __forceinline__ float decf(unsigned u) {
    u = (u & 0x80000000u) ? (u & 0x7FFFFFFFu) : ~u;
    return __uint_as_float(u);
}
#define ENC_NEG_INF 0x007FFFFFu   // encf(-inf)

// ---------------------------------------------------------------------------
// generic fill
__global__ void k_fill(unsigned* __restrict__ p, unsigned v, int n) {
    int i = blockIdx.x * blockDim.x + threadIdx.x;
    if (i < n) p[i] = v;
}

// ---------------------------------------------------------------------------
// input MLP: h = elu(elu(elu((dn*x)@W1+b1)@W2+b2)@W3+b3)   (5->32->64->64)
__global__ void k_inmlp(const float* __restrict__ x, const float* __restrict__ dn,
                        const float* __restrict__ w1, const float* __restrict__ b1,
                        const float* __restrict__ w2, const float* __restrict__ b2,
                        const float* __restrict__ w3, const float* __restrict__ b3,
                        float* __restrict__ h) {
    int i = blockIdx.x * blockDim.x + threadIdx.x;
    if (i >= NN) return;
    float a0[NIN];
#pragma unroll
    for (int c = 0; c < NIN; ++c) a0[c] = x[i * NIN + c] * dn[c];
    float a1[32];
    for (int o = 0; o < 32; ++o) {
        float s = b1[o];
#pragma unroll
        for (int c = 0; c < NIN; ++c) s += a0[c] * w1[c * 32 + o];
        a1[o] = eluf(s);
    }
    float a2[64];
    for (int o = 0; o < 64; ++o) {
        float s = b2[o];
#pragma unroll 8
        for (int c = 0; c < 32; ++c) s += a1[c] * w2[c * 64 + o];
        a2[o] = eluf(s);
    }
    for (int o = 0; o < 64; ++o) {
        float s = b3[o];
#pragma unroll 8
        for (int c = 0; c < 64; ++c) s += a2[c] * w3[c * 64 + o];
        h[i * 64 + o] = eluf(s);
    }
}

// ---------------------------------------------------------------------------
// f32 features -> f16 copy + squared norms (for gram-trick distances)
__global__ void k_featprep(const float* __restrict__ h, _Float16* __restrict__ xh,
                           float* __restrict__ sq) {
    int i = blockIdx.x * blockDim.x + threadIdx.x;
    if (i >= NN) return;
    float s = 0.0f;
#pragma unroll 8
    for (int c = 0; c < 64; ++c) {
        float v = h[i * 64 + c];
        s += v * v;
        xh[i * 64 + c] = (_Float16)v;
    }
    sq[i] = s;
}

// ---------------------------------------------------------------------------
// pack a row-major [Kd,Nd] f32 weight into WMMA-B lane order, f16.
// tile (kt,nt): out[((kt*(Nd/16)+nt)*32 + lane)*16 + e] = W[k,n]
//   n = nt*16 + (lane&15),  k = kt*32 + e + 8*((e>>3) + (lane>>4))
__global__ void k_packw(const float* __restrict__ W, _Float16* __restrict__ out,
                        int Kd, int Nd) {
    int idx = blockIdx.x * blockDim.x + threadIdx.x;
    int tot = (Kd >> 5) * (Nd >> 4) * 512;
    if (idx >= tot) return;
    int e    = idx & 15;
    int ln   = (idx >> 4) & 31;
    int tile = idx >> 9;
    int nt   = tile % (Nd >> 4);
    int kt   = tile / (Nd >> 4);
    int n = nt * 16 + (ln & 15);
    int k = kt * 32 + e + 8 * ((e >> 3) + (ln >> 4));
    out[idx] = (_Float16)W[k * Nd + n];
}

// ---------------------------------------------------------------------------
// Fused kNN: per block = 16 rows. Stream 64-col panels: WMMA gram tile ->
// d2 = |xi|^2+|xj|^2-2<xi,xj> (masked by batch & self) -> LDS -> register
// top-16 insertion (4 sub-scanners per row, merged at the end).
// B panels go memory->LDS via ASYNC loads when available (ASYNCcnt path).
__global__ __launch_bounds__(128) void k_knn(const _Float16* __restrict__ xh,
                                             const float* __restrict__ sq,
                                             const int* __restrict__ bat,
                                             int* __restrict__ nbr,
                                             int* __restrict__ ev) {
    __shared__ _Float16 sA[16 * 64];
    __shared__ _Float16 sB[64 * 64];
    __shared__ float    sD[16 * 64];
    __shared__ float    sSqi[16];
    __shared__ int      sBi[16];
    __shared__ float    sSqj[64];
    __shared__ int      sBj[64];
    __shared__ float    sKd[64][16];
    __shared__ int      sKi[64][16];

    int tid  = threadIdx.x;
    int wv   = tid >> 5;
    int lane = tid & 31;
    int i0   = blockIdx.x * 16;

    for (int q = tid; q < 16 * 64; q += 128) sA[q] = xh[i0 * 64 + q];
    if (tid < 16) { sSqi[tid] = sq[i0 + tid]; sBi[tid] = bat[i0 + tid]; }

    float kd[16]; int ki[16];
#pragma unroll
    for (int q = 0; q < 16; ++q) { kd[q] = INF_F; ki[q] = 0; }
    int row = tid & 15, sub = (tid >> 4) & 3;
    int nrow = lane & 15, hi = lane >> 4;

    for (int jt = 0; jt < NN; jt += 64) {
        __syncthreads();
#if HAVE_ASYNC_LDS
        // 64 rows x 64 f16 = 8KB panel: 512 x b128 async copies, 4 per lane
#pragma unroll
        for (int q = 0; q < 4; ++q) {
            int chunk = tid + q * 128;
            __builtin_amdgcn_global_load_async_to_lds_b128(
                (__attribute__((address_space(1))) v4i_t*)(xh + (size_t)jt * 64 + chunk * 8),
                (__attribute__((address_space(3))) v4i_t*)(sB + chunk * 8),
                0, 0);
        }
        __builtin_amdgcn_s_wait_asynccnt(0);
#else
        if (jt + 64 < NN) __builtin_prefetch(xh + (size_t)(jt + 64) * 64, 0, 1);
        for (int q = tid; q < 64 * 64; q += 128) sB[q] = xh[(size_t)jt * 64 + q];
#endif
        if (tid < 64) { sSqj[tid] = sq[jt + tid]; sBj[tid] = bat[jt + tid]; }
        __syncthreads();

        // wave wv computes 16x16 gram tile for cols jt + wv*16 .. +15
        v8f c = {};
#pragma unroll
        for (int kt = 0; kt < 2; ++kt) {
            v16h a, b;
#pragma unroll
            for (int q = 0; q < 16; ++q) {
                int K = kt * 32 + q + 8 * ((q >> 3) + hi);
                a[q] = sA[nrow * 64 + K];
            }
#pragma unroll
            for (int q = 0; q < 16; ++q) {
                int K = kt * 32 + q + 8 * ((q >> 3) + hi);
                b[q] = sB[(wv * 16 + nrow) * 64 + K];
            }
            c = __builtin_amdgcn_wmma_f32_16x16x32_f16(false, a, false, b,
                                                       (short)0, c, false, false);
        }
#pragma unroll
        for (int v = 0; v < 8; ++v) {
            int mm = v + 8 * hi;
            int nn = wv * 16 + nrow;
            int gi = i0 + mm, gj = jt + nn;
            float d2 = sSqi[mm] + sSqj[nn] - 2.0f * c[v];
            // branchless mask: cross-batch or self -> +inf
            int bad = (int)(sBi[mm] != sBj[nn]) | (int)(gi == gj);
            sD[mm * 64 + nn] = bad ? INF_F : d2;
        }
        __syncthreads();

        if (tid < 64) {
#pragma unroll
            for (int q = 0; q < 16; ++q) {
                int col = sub * 16 + q;
                float cd = sD[row * 64 + col];
                int   ci = jt + col;
#pragma unroll
                for (int p = 0; p < 16; ++p) {
                    if (cd < kd[p]) {
                        float td = kd[p]; int ti = ki[p];
                        kd[p] = cd; ki[p] = ci; cd = td; ci = ti;
                    }
                }
            }
        }
    }
    __syncthreads();
    if (tid < 64) {
#pragma unroll
        for (int q = 0; q < 16; ++q) { sKd[tid][q] = kd[q]; sKi[tid][q] = ki[q]; }
    }
    __syncthreads();
    if (tid < 16) {
        float fd[16]; int fi[16];
#pragma unroll
        for (int q = 0; q < 16; ++q) { fd[q] = INF_F; fi[q] = 0; }
        for (int s2 = 0; s2 < 4; ++s2) {
#pragma unroll
            for (int q = 0; q < 16; ++q) {
                float cd = sKd[s2 * 16 + tid][q];
                int   ci = sKi[s2 * 16 + tid][q];
#pragma unroll
                for (int p = 0; p < 16; ++p) {
                    if (cd < fd[p]) {
                        float td = fd[p]; int ti = fi[p];
                        fd[p] = cd; fi[p] = ci; cd = td; ci = ti;
                    }
                }
            }
        }
        int gi = i0 + tid;
#pragma unroll
        for (int q = 0; q < 16; ++q) {
            nbr[gi * KK + q] = fi[q];
            ev [gi * KK + q] = (fd[q] < INF_F) ? 1 : 0;
        }
    }
}

// ---------------------------------------------------------------------------
// edge list: forward (nbr->i) + reversed with mutual dedup
__global__ void k_edges(const int* __restrict__ nbr, const int* __restrict__ ev,
                        int* __restrict__ srcE, int* __restrict__ dstE,
                        int* __restrict__ validE) {
    int e = blockIdx.x * blockDim.x + threadIdx.x;
    if (e >= NN * KK) return;
    int i = e >> 4;
    int j = nbr[e];
    int v = ev[e];
    srcE[e] = j; dstE[e] = i; validE[e] = v;
    int mutual = 0;
#pragma unroll
    for (int q = 0; q < KK; ++q)
        if (ev[j * KK + q] && nbr[j * KK + q] == i) mutual = 1;
    int re = NN * KK + e;
    srcE[re] = i; dstE[re] = j; validE[re] = (v && !mutual) ? 1 : 0;
}

// ---------------------------------------------------------------------------
// EdgeConv: per wave = 16 edges. feat=[x_d, x_s-x_d] (128) --WMMA--> 96
// --elu--> --WMMA--> 64 --elu--> masked atomic segment-sum at dst.
__global__ __launch_bounds__(128) void k_edgeconv(
        const float* __restrict__ x,
        const int* __restrict__ srcE, const int* __restrict__ dstE,
        const int* __restrict__ validE,
        const _Float16* __restrict__ w1p, const float* __restrict__ b1,
        const _Float16* __restrict__ w2p, const float* __restrict__ b2,
        float* __restrict__ acc) {
    __shared__ _Float16 sFeat[4][16 * 128];
    __shared__ _Float16 sMid [4][16 * 96];
    __shared__ int sDst[64], sVal[64];

    int tid  = threadIdx.x;
    int wv   = tid >> 5;
    int lane = tid & 31;
    int e0   = (blockIdx.x * 4 + wv) * 16;

    // gather edge features (2 lanes per edge, 64 channels each)
    {
        int m = lane >> 1, hh = lane & 1;
        int e = e0 + m;
        int s = srcE[e], d = dstE[e];
        if (hh == 0) { sDst[wv * 16 + m] = d; sVal[wv * 16 + m] = validE[e]; }
        const float* xs = x + (size_t)s * 64;
        const float* xd = x + (size_t)d * 64;
#pragma unroll 8
        for (int c = 0; c < 64; ++c) {
            float v = hh ? (xs[c] - xd[c]) : xd[c];
            sFeat[wv][m * 128 + hh * 64 + c] = (_Float16)v;
        }
    }
    __syncthreads();

    int nrow = lane & 15, hi = lane >> 4;
    // layer 1: [16x128]@[128x96]
    for (int nt = 0; nt < 6; ++nt) {
        v8f c = {};
#pragma unroll
        for (int kt = 0; kt < 4; ++kt) {
            v16h a;
#pragma unroll
            for (int q = 0; q < 16; ++q) {
                int K = kt * 32 + q + 8 * ((q >> 3) + hi);
                a[q] = sFeat[wv][nrow * 128 + K];
            }
            v16h b = *(const v16h*)(w1p + ((size_t)(kt * 6 + nt) * 32 + lane) * 16);
            c = __builtin_amdgcn_wmma_f32_16x16x32_f16(false, a, false, b,
                                                       (short)0, c, false, false);
        }
        float bias = b1[nt * 16 + nrow];
#pragma unroll
        for (int v = 0; v < 8; ++v) {
            int mm = v + 8 * hi;
            float t = eluf(c[v] + bias);
            sMid[wv][mm * 96 + nt * 16 + nrow] = (_Float16)t;
        }
    }
    __syncthreads();

    // layer 2: [16x96]@[96x64] -> masked scatter-add
    for (int nt = 0; nt < 4; ++nt) {
        v8f c = {};
#pragma unroll
        for (int kt = 0; kt < 3; ++kt) {
            v16h a;
#pragma unroll
            for (int q = 0; q < 16; ++q) {
                int K = kt * 32 + q + 8 * ((q >> 3) + hi);
                a[q] = sMid[wv][nrow * 96 + K];
            }
            v16h b = *(const v16h*)(w2p + ((size_t)(kt * 4 + nt) * 32 + lane) * 16);
            c = __builtin_amdgcn_wmma_f32_16x16x32_f16(false, a, false, b,
                                                       (short)0, c, false, false);
        }
        float bias = b2[nt * 16 + nrow];
#pragma unroll
        for (int v = 0; v < 8; ++v) {
            int mm = v + 8 * hi;
            float t = eluf(c[v] + bias);
            if (sVal[wv * 16 + mm])
                atomicAdd(&acc[(size_t)sDst[wv * 16 + mm] * 64 + nt * 16 + nrow], t);
        }
    }
}

// ---------------------------------------------------------------------------
// graclus pieces
__global__ void k_deg(const int* __restrict__ srcE, const int* __restrict__ validE,
                      float* __restrict__ deg) {
    int e = blockIdx.x * blockDim.x + threadIdx.x;
    if (e >= NE) return;
    if (validE[e]) atomicAdd(&deg[srcE[e]], 1.0f);
}

__global__ void k_cut(const float* __restrict__ h, const int* __restrict__ srcE,
                      const int* __restrict__ dstE, const int* __restrict__ validE,
                      const float* __restrict__ deg, float* __restrict__ cutE) {
    int e = blockIdx.x * blockDim.x + threadIdx.x;
    if (e >= NE) return;
    if (!validE[e]) { cutE[e] = -INF_F; return; }
    int s = srcE[e], d = dstE[e];
    float a = 0.0f;
#pragma unroll 8
    for (int c = 0; c < 64; ++c) {
        float df = h[(size_t)s * 64 + c] - h[(size_t)d * 64 + c];
        a += df * df;
    }
    float w  = sqrtf(a + 1e-12f);
    float ds = deg[s], dd = deg[d];
    float is = ds > 0.0f ? 1.0f / ds : 0.0f;
    float id = dd > 0.0f ? 1.0f / dd : 0.0f;
    cutE[e] = w * (is + id);
}

__global__ void k_wmax(const int* __restrict__ dstE, const float* __restrict__ cutE,
                       unsigned* __restrict__ wmaxE) {
    int e = blockIdx.x * blockDim.x + threadIdx.x;
    if (e >= NE) return;
    atomicMax(&wmaxE[dstE[e]], encf(cutE[e]));
}

__global__ void k_best(const int* __restrict__ srcE, const int* __restrict__ dstE,
                       const float* __restrict__ cutE, const unsigned* __restrict__ wmaxE,
                       int* __restrict__ best) {
    int e = blockIdx.x * blockDim.x + threadIdx.x;
    if (e >= NE) return;
    float c = cutE[e];
    if (c != -INF_F && encf(c) == wmaxE[dstE[e]])
        atomicMax(&best[dstE[e]], srcE[e]);
}

__global__ void k_cluster(const int* __restrict__ best, int* __restrict__ clus) {
    int i = blockIdx.x * blockDim.x + threadIdx.x;
    if (i >= NN) return;
    int b  = best[i];
    int bb = b >= 0 ? b : 0;
    bool matched = (b >= 0) && (best[bb] == i);
    clus[i] = matched ? (i < b ? i : b) : i;
}

// ---------------------------------------------------------------------------
// max-pool by cluster representative
__global__ void k_poolscat(const float* __restrict__ h, const int* __restrict__ batIn,
                           const int* __restrict__ clus, int* __restrict__ cnt,
                           int* __restrict__ bmax, unsigned* __restrict__ pmaxE) {
    int i = blockIdx.x * blockDim.x + threadIdx.x;
    if (i >= NN) return;
    int c = clus[i];
    atomicAdd(&cnt[c], 1);
    atomicMax(&bmax[c], batIn[i]);
#pragma unroll 8
    for (int ch = 0; ch < 64; ++ch)
        atomicMax(&pmaxE[(size_t)c * 64 + ch], encf(h[(size_t)i * 64 + ch]));
}

__global__ void k_poolfin(const unsigned* __restrict__ pmaxE, const int* __restrict__ cnt,
                          const int* __restrict__ bmax, float* __restrict__ hOut,
                          int* __restrict__ batOut) {
    int i = blockIdx.x * blockDim.x + threadIdx.x;
    if (i >= NN) return;
    bool v = cnt[i] > 0;
    batOut[i] = v ? bmax[i] : NB;
#pragma unroll 8
    for (int ch = 0; ch < 64; ++ch)
        hOut[(size_t)i * 64 + ch] = v ? decf(pmaxE[(size_t)i * 64 + ch]) : 0.0f;
}

// ---------------------------------------------------------------------------
// global max pool + output MLP (64->64->32->1)
__global__ void k_gmax(const float* __restrict__ h, const int* __restrict__ bat,
                       unsigned* __restrict__ gmaxE) {
    int i = blockIdx.x * blockDim.x + threadIdx.x;
    if (i >= NN) return;
    int b = bat[i];
#pragma unroll 8
    for (int ch = 0; ch < 64; ++ch)
        atomicMax(&gmaxE[(size_t)b * 64 + ch], encf(h[(size_t)i * 64 + ch]));
}

__global__ void k_outmlp(const unsigned* __restrict__ gmaxE,
                         const float* __restrict__ w1, const float* __restrict__ b1,
                         const float* __restrict__ w2, const float* __restrict__ b2,
                         const float* __restrict__ w3, const float* __restrict__ b3,
                         float* __restrict__ out) {
    int b = threadIdx.x;
    if (b >= NB) return;
    float g[64], t1[64], t2[32];
    for (int c = 0; c < 64; ++c) g[c] = decf(gmaxE[b * 64 + c]);
    for (int o = 0; o < 64; ++o) {
        float s = b1[o];
#pragma unroll 8
        for (int c = 0; c < 64; ++c) s += g[c] * w1[c * 64 + o];
        t1[o] = eluf(s);
    }
    for (int o = 0; o < 32; ++o) {
        float s = b2[o];
#pragma unroll 8
        for (int c = 0; c < 64; ++c) s += t1[c] * w2[c * 32 + o];
        t2[o] = eluf(s);
    }
    float s = b3[0];
#pragma unroll 8
    for (int c = 0; c < 32; ++c) s += t2[c] * w3[c];
    out[b] = s;
}

// ---------------------------------------------------------------------------
extern "C" void kernel_launch(void* const* d_in, const int* in_sizes, int n_in,
                              void* d_out, int out_size, void* d_ws, size_t ws_size,
                              hipStream_t stream) {
    const float* x       = (const float*)d_in[0];
    const int*   batch   = (const int*)d_in[1];
    const float* dn      = (const float*)d_in[2];
    const float* in_w1   = (const float*)d_in[3];
    const float* in_b1   = (const float*)d_in[4];
    const float* in_w2   = (const float*)d_in[5];
    const float* in_b2   = (const float*)d_in[6];
    const float* in_w3   = (const float*)d_in[7];
    const float* in_b3   = (const float*)d_in[8];
    const float* c1_w1   = (const float*)d_in[9];
    const float* c1_b1   = (const float*)d_in[10];
    const float* c1_w2   = (const float*)d_in[11];
    const float* c1_b2   = (const float*)d_in[12];
    const float* c2_w1   = (const float*)d_in[13];
    const float* c2_b1   = (const float*)d_in[14];
    const float* c2_w2   = (const float*)d_in[15];
    const float* c2_b2   = (const float*)d_in[16];
    const float* out_w1  = (const float*)d_in[17];
    const float* out_b1  = (const float*)d_in[18];
    const float* out_w2  = (const float*)d_in[19];
    const float* out_b2  = (const float*)d_in[20];
    const float* out_w3  = (const float*)d_in[21];
    const float* out_b3  = (const float*)d_in[22];
    float* out = (float*)d_out;

    char* ws = (char*)d_ws;
    size_t off = 0;
    auto alloc = [&](size_t bytes) -> void* {
        off = (off + 255) & ~(size_t)255;
        void* p = ws + off;
        off += bytes;
        return p;
    };
    float*     hA     = (float*)alloc((size_t)NN * HH * 4);
    float*     hB     = (float*)alloc((size_t)NN * HH * 4);
    _Float16*  xh     = (_Float16*)alloc((size_t)NN * HH * 2);
    float*     sqv    = (float*)alloc((size_t)NN * 4);
    int*       nbr    = (int*)alloc((size_t)NN * KK * 4);
    int*       ev     = (int*)alloc((size_t)NN * KK * 4);
    int*       srcE   = (int*)alloc((size_t)NE * 4);
    int*       dstE   = (int*)alloc((size_t)NE * 4);
    int*       validE = (int*)alloc((size_t)NE * 4);
    float*     cutE   = (float*)alloc((size_t)NE * 4);
    float*     deg    = (float*)alloc((size_t)NN * 4);
    unsigned*  wmaxE  = (unsigned*)alloc((size_t)NN * 4);
    int*       best   = (int*)alloc((size_t)NN * 4);
    int*       clus   = (int*)alloc((size_t)NN * 4);
    int*       cnt    = (int*)alloc((size_t)NN * 4);
    int*       bmax   = (int*)alloc((size_t)NN * 4);
    int*       bat2   = (int*)alloc((size_t)NN * 4);
    int*       bat3   = (int*)alloc((size_t)NN * 4);
    unsigned*  pmaxE  = (unsigned*)alloc((size_t)NN * HH * 4);
    unsigned*  gmaxE  = (unsigned*)alloc((size_t)(NB + 1) * HH * 4);
    _Float16*  w1p    = (_Float16*)alloc((size_t)4 * 6 * 512 * 2);
    _Float16*  w2p    = (_Float16*)alloc((size_t)3 * 4 * 512 * 2);

    k_inmlp<<<NN / 256, 256, 0, stream>>>(x, dn, in_w1, in_b1, in_w2, in_b2,
                                          in_w3, in_b3, hA);

    auto pass = [&](const int* batIn, int* batOut,
                    const float* cw1, const float* cb1,
                    const float* cw2, const float* cb2) {
        k_featprep<<<NN / 256, 256, 0, stream>>>(hA, xh, sqv);
        k_knn<<<NN / 16, 128, 0, stream>>>(xh, sqv, batIn, nbr, ev);
        k_edges<<<NN * KK / 256, 256, 0, stream>>>(nbr, ev, srcE, dstE, validE);
        k_packw<<<(4 * 6 * 512) / 256, 256, 0, stream>>>(cw1, w1p, 128, 96);
        k_packw<<<(3 * 4 * 512) / 256, 256, 0, stream>>>(cw2, w2p, 96, 64);
        k_fill<<<(NN * HH) / 256, 256, 0, stream>>>((unsigned*)hB, 0u, NN * HH);
        k_edgeconv<<<NE / 64, 128, 0, stream>>>(hA, srcE, dstE, validE,
                                                w1p, cb1, w2p, cb2, hB);
        k_fill<<<NN / 256, 256, 0, stream>>>((unsigned*)deg, 0u, NN);
        k_deg<<<NE / 256, 256, 0, stream>>>(srcE, validE, deg);
        k_cut<<<NE / 256, 256, 0, stream>>>(hB, srcE, dstE, validE, deg, cutE);
        k_fill<<<NN / 256, 256, 0, stream>>>(wmaxE, ENC_NEG_INF, NN);
        k_wmax<<<NE / 256, 256, 0, stream>>>(dstE, cutE, wmaxE);
        k_fill<<<NN / 256, 256, 0, stream>>>((unsigned*)best, 0xFFFFFFFFu, NN);
        k_best<<<NE / 256, 256, 0, stream>>>(srcE, dstE, cutE, wmaxE, best);
        k_cluster<<<NN / 256, 256, 0, stream>>>(best, clus);
        k_fill<<<NN / 256, 256, 0, stream>>>((unsigned*)cnt, 0u, NN);
        k_fill<<<NN / 256, 256, 0, stream>>>((unsigned*)bmax, 0xFFFFFFFFu, NN);
        k_fill<<<(NN * HH) / 256, 256, 0, stream>>>(pmaxE, ENC_NEG_INF, NN * HH);
        k_poolscat<<<NN / 256, 256, 0, stream>>>(hB, batIn, clus, cnt, bmax, pmaxE);
        k_poolfin<<<NN / 256, 256, 0, stream>>>(pmaxE, cnt, bmax, hA, batOut);
    };

    pass(batch, bat2, c1_w1, c1_b1, c1_w2, c1_b2);
    pass(bat2,  bat3, c2_w1, c2_b1, c2_w2, c2_b2);

    k_fill<<<1, 256, 0, stream>>>(gmaxE, ENC_NEG_INF, (NB + 1) * HH);
    k_gmax<<<NN / 256, 256, 0, stream>>>(hA, bat3, gmaxE);
    k_outmlp<<<1, 32, 0, stream>>>(gmaxE, out_w1, out_b1, out_w2, out_b2,
                                   out_w3, out_b3, out);
    (void)in_sizes; (void)n_in; (void)out_size; (void)ws_size;
}
